// CrossWindowAttention_49203145343523
// MI455X (gfx1250) — matrix operand order repaired and checked by reading, not compiled, not run on hardware
//
#include <hip/hip_runtime.h>

// Problem constants (from reference): B=1024, N=64, NKV=128, DIM=256, HEADS=8, HD=32
#define NB    1024
#define NQ    64
#define NKV   128
#define DIM   256
#define HEADS 8
#define HD    32
#define SCALE 0.17677669529663687f   // 32^-0.5
#define COS_EPS 1e-7f

typedef __attribute__((ext_vector_type(16))) _Float16 v16h;
typedef __attribute__((ext_vector_type(8)))  _Float16 h8;
typedef __attribute__((ext_vector_type(4)))  _Float16 h4;
typedef __attribute__((ext_vector_type(8)))  float    v8f;

union FragH { v16h v; h8 c[2]; _Float16 h[16]; };

// A-fragment (16x32 f16, M x K), source row-major with leading dim ld (halfs).
// Per ISA: lane L holds row M=L%16; VGPR v<4 -> K = k0 + (L/16)*8 + 2v+j ;
//          VGPR v>=4 -> K = k0 + 16 + (L/16)*8 + 2(v-4)+j  => two contiguous 8-half chunks.
__device__ __forceinline__ v16h load_a_frag(const _Float16* base, int m0, int k0, int ld, int lane) {
  const int row = m0 + (lane & 15);
  const int hh  = lane >> 4;
  const _Float16* p = base + row * ld + k0 + hh * 8;
  FragH f;
  f.c[0] = *(const h8*)(p);
  f.c[1] = *(const h8*)(p + 16);
  return f.v;
}

// B-fragment (32x16 f16, K x N) where source is stored TRANSPOSED (N-major, T[n][k], ld in halfs).
// Per ISA: lane L holds col N=L%16; same K order as A: two contiguous 8-half chunks at k0+(L/16)*8 and +16.
__device__ __forceinline__ v16h load_bT_frag(const _Float16* baseT, int n0, int k0, int ld, int lane) {
  const int row = n0 + (lane & 15);
  const int hh  = lane >> 4;
  const _Float16* p = baseT + row * ld + k0 + hh * 8;
  FragH f;
  f.c[0] = *(const h8*)(p);
  f.c[1] = *(const h8*)(p + 16);
  return f.v;
}

#define WMMA(a, b, c) __builtin_amdgcn_wmma_f32_16x16x32_f16(false, (a), false, (b), (short)0, (c), false, false)
#define V8F_ZERO {0.f,0.f,0.f,0.f,0.f,0.f,0.f,0.f}

// ---- prep: transpose weights to f16 N-major layouts in workspace ----
// ws halfs layout: WqT[256][256] @0, WkvT[512][256] @65536, WprojT[256][256] @196608
__global__ __launch_bounds__(256) void prep_weights(
    const float* __restrict__ Wq, const float* __restrict__ Wkv,
    const float* __restrict__ Wproj, _Float16* __restrict__ wsW) {
  int i = blockIdx.x * 256 + threadIdx.x;
  if (i < 65536) {
    int n = i >> 8, k = i & 255;
    wsW[i] = (_Float16)Wq[k * 256 + n];
  } else if (i < 65536 + 131072) {
    int j = i - 65536; int n = j >> 8, k = j & 255;
    wsW[i] = (_Float16)Wkv[k * 512 + n];
  } else if (i < 262144) {
    int j = i - 196608; int n = j >> 8, k = j & 255;
    wsW[i] = (_Float16)Wproj[k * 256 + n];
  }
}

// Per-head rank-1 mult_mask constants: hc[w*8+h], w: 0=Wpq.Wps 1=Wpq.bps 2=bpq.Wps 3=bpq.bps
__global__ void prep_headc(const float* __restrict__ Wpq, const float* __restrict__ bpq,
                           const float* __restrict__ Wps, const float* __restrict__ bps,
                           float* __restrict__ hc) {
  int t = threadIdx.x;
  if (t >= 32) return;
  int h = t >> 2, w = t & 3;
  const float* a = (w & 2) ? bpq : Wpq;
  const float* c = (w & 1) ? bps : Wps;
  float sacc = 0.f;
  for (int j = 0; j < HD; ++j) sacc += a[h * HD + j] * c[h * HD + j];
  hc[w * 8 + h] = sacc;
}

// ---- main fused kernel: one workgroup (8 waves) per batch ----
// Dynamic LDS map (bytes):
//   [0,      65536)  kL   f16 [128][256]  (k, row-major: serves as B^T for q.k^T)
//   [65536, 131072)  vT   f16 [256][128]  (v transposed: serves as B^T for P.V)
//   [131072,163840)  qhL  f16 [64][256]
//   [163840,196608)  oL   f16 [64][256]
//   [196608,262144)  attn f32 [2][64][128] (per head-group; P f16 written in place)
//   [262144,267264)  qnrm[512] knrm[512] qm[64] sm[64] pp[64] hor[64]
//   staging (phase 1 only, overlays attn+small): qS f16[64][256] @196608, sS f16[128][256] @229376
#define LDS_BYTES 294912

__global__ __launch_bounds__(256) void cwa_kernel(
    const float* __restrict__ qg, const float* __restrict__ sg,
    const float* __restrict__ q_mask, const float* __restrict__ s_mask,
    const float* __restrict__ pro_pred,
    const float* __restrict__ bq, const float* __restrict__ bkv,
    const float* __restrict__ bproj,
    const _Float16* __restrict__ WqT, const _Float16* __restrict__ WkvT,
    const _Float16* __restrict__ WprojT,
    const float* __restrict__ hc,
    float* __restrict__ out) {
  extern __shared__ __align__(16) char smem[];
  _Float16* kL  = (_Float16*)smem;                    // [128][256]
  _Float16* vT  = kL + 128 * 256;                     // [256][128]
  _Float16* qhL = vT + 256 * 128;                     // [64][256]
  _Float16* oL  = qhL + 64 * 256;                     // [64][256]
  float*    attnF = (float*)(smem + 196608);          // [2][64][128]
  float*    qnrm  = (float*)(smem + 262144);          // [512]
  float*    knrm  = qnrm + 512;                       // [512]
  float*    qmv   = knrm + 512;                       // [64]
  float*    smv   = qmv + 64;                         // [64]
  float*    ppv   = smv + 64;                         // [64]
  float*    horv  = ppv + 64;                         // [64]
  _Float16* qS = (_Float16*)(smem + 196608);          // staging [64][256]
  _Float16* sS = qS + 64 * 256;                       // staging [128][256]

  const int tid  = threadIdx.x;
  const int wave = tid >> 5;
  const int lane = tid & 31;
  const int b    = blockIdx.x;

  // ---- Phase 1: stage q, s as f16 into LDS ----
  {
    const float4* qp = (const float4*)(qg + (size_t)b * NQ * DIM);
    for (int i = tid; i < NQ * DIM / 4; i += 256) {
      float4 x = qp[i];
      h4 y = {(_Float16)x.x, (_Float16)x.y, (_Float16)x.z, (_Float16)x.w};
      *(h4*)(qS + i * 4) = y;
    }
    const float4* sp = (const float4*)(sg + (size_t)b * NKV * DIM);
    for (int i = tid; i < NKV * DIM / 4; i += 256) {
      float4 x = sp[i];
      h4 y = {(_Float16)x.x, (_Float16)x.y, (_Float16)x.z, (_Float16)x.w};
      *(h4*)(sS + i * 4) = y;
    }
  }
  __syncthreads();

  // ---- Phase 2: qh = q @ Wq + bq  (64x256x256) ----
  for (int t = wave; t < 64; t += 8) {
    const int m0 = (t & 3) * 16, n0 = (t >> 2) * 16;
    v8f acc = V8F_ZERO;
#pragma unroll
    for (int k0 = 0; k0 < DIM; k0 += 32) {
      v16h a  = load_a_frag(qS, m0, k0, DIM, lane);
      v16h bb = load_bT_frag(WqT, n0, k0, DIM, lane);
      acc = WMMA(a, bb, acc);
    }
    const int col = n0 + (lane & 15);
    const int rb  = m0 + ((lane >> 4) << 3);
    const float bias = bq[col];
#pragma unroll
    for (int r = 0; r < 8; ++r) qhL[(rb + r) * DIM + col] = (_Float16)(acc[r] + bias);
  }

  // ---- Phase 3: kv = s @ Wkv + bkv (128x256x512); k row-major, v transposed ----
  for (int t = wave; t < 256; t += 8) {
    const int m0 = (t & 7) * 16, c0 = (t >> 3) * 16;
    v8f acc = V8F_ZERO;
#pragma unroll
    for (int k0 = 0; k0 < DIM; k0 += 32) {
      v16h a  = load_a_frag(sS, m0, k0, DIM, lane);
      v16h bb = load_bT_frag(WkvT, c0, k0, DIM, lane);
      acc = WMMA(a, bb, acc);
    }
    const int c  = c0 + (lane & 15);
    const int rb = m0 + ((lane >> 4) << 3);
    const float bias = bkv[c];
    if (c0 < 256) {  // k-half
#pragma unroll
      for (int r = 0; r < 8; ++r) kL[(rb + r) * DIM + c] = (_Float16)(acc[r] + bias);
    } else {         // v-half, stored transposed: vT[dim][key]
      const int d = c - 256;
#pragma unroll
      for (int r = 0; r < 8; ++r) vT[d * NKV + (rb + r)] = (_Float16)(acc[r] + bias);
    }
  }
  __syncthreads();

  // ---- Phase 4: masks + q/k norms ----
  if (tid < 64) {
    const int n = tid;
    qmv[n] = (q_mask[b * NQ + n] >= 0.5f) ? 1.f : 0.f;
    smv[n] = (s_mask[b * NQ + n] >= 0.5f) ? 1.f : 0.f;
    float pr = pro_pred[b * NQ + n];
    float pp = (pr >= 1.f) ? 1.f : pr;
    ppv[n]  = pp;
    horv[n] = (pp < 0.3f) ? 0.f : ((pp > 0.7f) ? 1.f : pp);
  }
  for (int i = tid; i < 512; i += 256) {
    const int h = i >> 6, n = i & 63;
    float a = 0.f;
    const _Float16* pq = qhL + n * DIM + h * HD;
#pragma unroll
    for (int j = 0; j < HD; ++j) { float x = (float)pq[j]; a += x * x; }
    qnrm[i] = sqrtf(a);
    a = 0.f;
    const _Float16* pk = kL + (NQ + n) * DIM + h * HD;
#pragma unroll
    for (int j = 0; j < HD; ++j) { float x = (float)pk[j]; a += x * x; }
    knrm[i] = sqrtf(a);
  }
  __syncthreads();

  // ---- Phase 5: attention, 2 heads at a time ----
  for (int hg = 0; hg < 4; ++hg) {
    // 5a: logits  attn[g][n][key] = qh . k^T  (K = HD = 32 -> one WMMA per tile)
    for (int t = wave; t < 64; t += 8) {
      const int g = t >> 5, r5 = t & 31;
      const int m0 = (r5 & 3) * 16, n0 = (r5 >> 2) * 16;
      const int h  = hg * 2 + g;
      v16h a  = load_a_frag(qhL, m0, h * HD, DIM, lane);
      v16h bb = load_bT_frag(kL, n0, h * HD, DIM, lane);   // kL row-major == (k^T) transposed
      v8f acc = V8F_ZERO;
      acc = WMMA(a, bb, acc);
      const int col = n0 + (lane & 15);
      const int rb  = m0 + ((lane >> 4) << 3);
      float* dst = attnF + g * (NQ * NKV);
#pragma unroll
      for (int r = 0; r < 8; ++r) dst[(rb + r) * NKV + col] = acc[r];
    }
    __syncthreads();

    // 5b: masks + cosine cross + softmax; write P f16 in place
    if (tid < 128) {
      const int g = tid >> 6, n = tid & 63;
      const int h = hg * 2 + g;
      float* row = attnF + g * (NQ * NKV) + n * NKV;
      const float qn = qnrm[h * 64 + n];
      const float ppn = ppv[n], horn = horv[n], qmn = qmv[n];
      const float cww = hc[h], cwb = hc[8 + h], cbw = hc[16 + h], cbb = hc[24 + h];
      float mx = -1e30f;
      for (int m = 0; m < 64; ++m) {            // self block
        float a = row[m] * SCALE;
        if (qmn + qmv[m] == 1.0f) a = -1e30f;
        row[m] = a;
        mx = fmaxf(mx, a);
      }
      for (int m = 0; m < 64; ++m) {            // cross block
        float c   = row[64 + m];
        float den = qn * knrm[h * 64 + m] + COS_EPS;
        float smm = smv[m];
        float mm  = ppn * smm * cww + ppn * cwb + smm * cbw + cbb;
        float a   = mm * (c / den);
        if (horn + smm == 1.0f) a = -1e30f;
        row[64 + m] = a;
        mx = fmaxf(mx, a);
      }
      float sum = 0.f;
      for (int m = 0; m < NKV; ++m) sum += __expf(row[m] - mx);
      const float inv = 1.0f / sum;
      _Float16* P = (_Float16*)row;             // in-place f32 -> f16 (ascending is safe)
      for (int m = 0; m < NKV; ++m) {
        float p = __expf(row[m] - mx) * inv;
        P[m] = (_Float16)p;
      }
    }
    __syncthreads();

    // 5c: out_h = P @ V   (64 x 32, K = 128); P rows have ld = 256 halfs (f32 row stride)
    for (int t = wave; t < 16; t += 8) {
      const int g = t >> 3, r3 = t & 7;
      const int m0 = (r3 >> 1) * 16, n0v = (r3 & 1) * 16;
      const int h  = hg * 2 + g;
      const _Float16* P = (const _Float16*)(attnF + g * (NQ * NKV));
      v8f acc = V8F_ZERO;
#pragma unroll
      for (int k0 = 0; k0 < NKV; k0 += 32) {
        v16h a  = load_a_frag(P, m0, k0, 2 * NKV, lane);
        v16h bb = load_bT_frag(vT, h * HD + n0v, k0, NKV, lane);
        acc = WMMA(a, bb, acc);
      }
      const int col = h * HD + n0v + (lane & 15);
      const int rb  = m0 + ((lane >> 4) << 3);
#pragma unroll
      for (int r = 0; r < 8; ++r) oL[(rb + r) * DIM + col] = (_Float16)acc[r];
    }
    __syncthreads();
  }

  // ---- Phase 6: out = o @ Wproj + bproj -> global f32 ----
  float* og = out + (size_t)b * NQ * DIM;
  for (int t = wave; t < 64; t += 8) {
    const int m0 = (t & 3) * 16, n0 = (t >> 2) * 16;
    v8f acc = V8F_ZERO;
#pragma unroll
    for (int k0 = 0; k0 < DIM; k0 += 32) {
      v16h a  = load_a_frag(oL, m0, k0, DIM, lane);
      v16h bb = load_bT_frag(WprojT, n0, k0, DIM, lane);
      acc = WMMA(a, bb, acc);
    }
    const int col = n0 + (lane & 15);
    const int rb  = m0 + ((lane >> 4) << 3);
    const float bias = bproj[col];
#pragma unroll
    for (int r = 0; r < 8; ++r) og[(rb + r) * DIM + col] = acc[r] + bias;
  }
}

extern "C" void kernel_launch(void* const* d_in, const int* in_sizes, int n_in,
                              void* d_out, int out_size, void* d_ws, size_t ws_size,
                              hipStream_t stream) {
  (void)in_sizes; (void)n_in; (void)out_size; (void)ws_size;
  const float* q        = (const float*)d_in[0];
  const float* s        = (const float*)d_in[1];
  const float* q_mask   = (const float*)d_in[2];
  const float* s_mask   = (const float*)d_in[3];
  const float* pro_pred = (const float*)d_in[4];
  const float* Wq       = (const float*)d_in[5];
  const float* bq       = (const float*)d_in[6];
  const float* Wkv      = (const float*)d_in[7];
  const float* bkv      = (const float*)d_in[8];
  const float* Wproj    = (const float*)d_in[9];
  const float* bproj    = (const float*)d_in[10];
  const float* Wpq      = (const float*)d_in[11];
  const float* bpq      = (const float*)d_in[12];
  const float* Wps      = (const float*)d_in[13];
  const float* bps      = (const float*)d_in[14];

  _Float16* wsW = (_Float16*)d_ws;                       // 262144 halfs = 512 KB
  float*    hc  = (float*)((char*)d_ws + 524288);        // 32 floats

  prep_weights<<<1024, 256, 0, stream>>>(Wq, Wkv, Wproj, wsW);
  prep_headc<<<1, 32, 0, stream>>>(Wpq, bpq, Wps, bps, hc);

  cwa_kernel<<<NB, 256, LDS_BYTES, stream>>>(
      q, s, q_mask, s_mask, pro_pred, bq, bkv, bproj,
      wsW, wsW + 65536, wsW + 196608, hc, (float*)d_out);
}
